// Block_63977832841590
// MI455X (gfx1250) — matrix operand order repaired and checked by reading, not compiled, not run on hardware
//
#include <hip/hip_runtime.h>
#include <math.h>

#define NN  4096
#define EE  512
#define HH  8
#define DD  64
#define FFD 2048

typedef __bf16 bf16;
typedef __attribute__((ext_vector_type(16))) __bf16 v16bf;
typedef __attribute__((ext_vector_type(8)))  __bf16 v8bf;
typedef __attribute__((ext_vector_type(4)))  __bf16 v4bf;
typedef __attribute__((ext_vector_type(8)))  float  v8f;
typedef __attribute__((ext_vector_type(4)))  float  v4f;

union Frag16 { v16bf v; v8bf h[2]; };

static __device__ __forceinline__ v8f wmma_bf16(v16bf a, v16bf b, v8f c) {
  // D = A(16x32 bf16) x B(32x16 bf16) + C(16x16 f32)
  return __builtin_amdgcn_wmma_f32_16x16x32_bf16(
      /*neg_a=*/false, a, /*neg_b=*/false, b,
      /*c_mod=*/(short)0, c, /*reuse_a=*/false, /*reuse_b=*/false);
}

// ---------------------------------------------------------------------------
// f32 -> bf16 flat conversion (vectorized x4)
// ---------------------------------------------------------------------------
__global__ __launch_bounds__(256) void k_f32_to_bf16(const float* __restrict__ in,
                                                     bf16* __restrict__ out, long n4) {
  long i = (long)blockIdx.x * 256 + threadIdx.x;
  if (i >= n4) return;
  v4f x = ((const v4f*)in)[i];
  v4bf o = {(bf16)x.x, (bf16)x.y, (bf16)x.z, (bf16)x.w};
  ((v4bf*)out)[i] = o;
}

// ---------------------------------------------------------------------------
// RMSNorm over E=512, one wave per row, output bf16
// ---------------------------------------------------------------------------
__global__ __launch_bounds__(256) void k_rmsnorm(const float* __restrict__ x,
                                                 const float* __restrict__ w,
                                                 bf16* __restrict__ out) {
  const int wave = threadIdx.x >> 5;
  const int lane = threadIdx.x & 31;
  const int row  = blockIdx.x * 8 + wave;
  const float* xr = x + (long)row * EE;
  float vals[16];
  float s = 0.0f;
#pragma unroll
  for (int i = 0; i < 16; ++i) {
    vals[i] = xr[lane + i * 32];
    s += vals[i] * vals[i];
  }
#pragma unroll
  for (int off = 16; off >= 1; off >>= 1) s += __shfl_xor(s, off, 32);
  const float scale = rsqrtf(s * (1.0f / (float)EE) + 1.1920928955078125e-07f);
  bf16* orow = out + (long)row * EE;
#pragma unroll
  for (int i = 0; i < 16; ++i) {
    int c = lane + i * 32;
    orow[c] = (bf16)(vals[i] * scale * w[c]);
  }
}

// ---------------------------------------------------------------------------
// WMMA GEMM:  C[M,Nc] = A[M,K](bf16,rm) @ Bt[Nc,K](bf16,rm)^T
// Block = 128 threads (4 waves, 2x2), wave tile 32x64 (2x4 WMMA tiles),
// block tile 64x128. Per K=32 step: 6 fragment loads, 8 WMMAs (~21 flop/B).
// Epilogue: optional bias (per column), exact GELU, f32 out, bf16 out,
// bf16 TRANSPOSED out (Ct[Nc, M], ldct) -- one b128 store per lane per tile.
// ---------------------------------------------------------------------------
template <bool BIAS, bool GELU, bool WF32, bool WBF16, bool WBF16T>
__global__ __launch_bounds__(128) void k_gemm_bt(
    const bf16* __restrict__ A, int lda,
    const bf16* __restrict__ Bt, int ldb,
    int K,
    const float* __restrict__ bias,
    float* __restrict__ Cf, bf16* __restrict__ Cb, int ldc,
    bf16* __restrict__ Ct, int ldct) {
  const int wave = threadIdx.x >> 5;
  const int lane = threadIdx.x & 31;
  const int lr = lane & 15;   // row within A tile / col within B tile
  const int lh = lane >> 4;   // half select
  const int wm = blockIdx.y * 64 + (wave >> 1) * 32;
  const int wn = blockIdx.x * 128 + (wave & 1) * 64;

  v8f acc[2][4];
#pragma unroll
  for (int i = 0; i < 2; ++i)
#pragma unroll
    for (int j = 0; j < 4; ++j) {
      v8f z = {0, 0, 0, 0, 0, 0, 0, 0};
      acc[i][j] = z;
    }

  // A fragment (16x32): lane holds row (lr), elems 0-15 = K[lh*16 .. +15]
  const bf16* Arow[2];
  Arow[0] = A + (long)(wm + lr) * lda + lh * 16;
  Arow[1] = Arow[0] + (long)16 * lda;
  // B fragment (32x16): lane holds col (lr), elems 0-15 = K[lh*16 .. +15] of Bt row
  const bf16* Brow[4];
  Brow[0] = Bt + (long)(wn + lr) * ldb + lh * 16;
  Brow[1] = Brow[0] + (long)16 * ldb;
  Brow[2] = Brow[0] + (long)32 * ldb;
  Brow[3] = Brow[0] + (long)48 * ldb;

  for (int k0 = 0; k0 < K; k0 += 32) {
    Frag16 a[2], b[4];
#pragma unroll
    for (int i = 0; i < 2; ++i) {
      a[i].h[0] = *(const v8bf*)(Arow[i] + k0);
      a[i].h[1] = *(const v8bf*)(Arow[i] + k0 + 8);
    }
#pragma unroll
    for (int j = 0; j < 4; ++j) {
      b[j].h[0] = *(const v8bf*)(Brow[j] + k0);
      b[j].h[1] = *(const v8bf*)(Brow[j] + k0 + 8);
    }
#pragma unroll
    for (int i = 0; i < 2; ++i)
#pragma unroll
      for (int j = 0; j < 4; ++j)
        acc[i][j] = wmma_bf16(a[i].v, b[j].v, acc[i][j]);
  }

  auto emit = [&](v8f c, int tm, int tn) {
    const int col = wn + tn + lr;
    const int rbase = wm + tm + lh * 8;
    float bv = 0.0f;
    if constexpr (BIAS) bv = bias[col];
    v8bf tv;
#pragma unroll
    for (int r = 0; r < 8; ++r) {
      float val = c[r] + bv;
      if constexpr (GELU)
        val = 0.5f * val * (1.0f + erff(val * 0.7071067811865475f));
      if constexpr (WF32)  Cf[(long)(rbase + r) * ldc + col] = val;
      if constexpr (WBF16) Cb[(long)(rbase + r) * ldc + col] = (bf16)val;
      tv[r] = (bf16)val;
    }
    if constexpr (WBF16T)
      *(v8bf*)(Ct + (long)col * ldct + rbase) = tv;  // Ct[col, rbase..rbase+7]
  };
#pragma unroll
  for (int i = 0; i < 2; ++i)
#pragma unroll
    for (int j = 0; j < 4; ++j)
      emit(acc[i][j], i * 16, j * 16);
}

// ---------------------------------------------------------------------------
// Per-node 8x8 head attention (tiny): one thread per (node, head).
// s = (q k^T) / sqrt(64); softmax over 8; out = a @ v. Optionally accumulate.
// ---------------------------------------------------------------------------
__global__ __launch_bounds__(256) void k_sdpa(const float* __restrict__ q,
                                              const float* __restrict__ k,
                                              const float* __restrict__ v,
                                              float* __restrict__ out, int accum) {
  const int t = blockIdx.x * 256 + threadIdx.x;  // t < N*H
  const int n = t >> 3;
  const int h = t & 7;
  const float* qp = q + (long)n * EE + h * DD;
  float qv[DD];
#pragma unroll
  for (int d = 0; d < DD; ++d) qv[d] = qp[d];
  float s[HH];
  float mx = -3.4e38f;
#pragma unroll
  for (int g = 0; g < HH; ++g) {
    const float* kp = k + (long)n * EE + g * DD;
    float a = 0.0f;
#pragma unroll
    for (int d = 0; d < DD; ++d) a += qv[d] * kp[d];
    a *= 0.125f;  // D^-0.5 = 1/8
    s[g] = a;
    mx = fmaxf(mx, a);
  }
  float sum = 0.0f;
#pragma unroll
  for (int g = 0; g < HH; ++g) { s[g] = expf(s[g] - mx); sum += s[g]; }
  const float inv = 1.0f / sum;
#pragma unroll
  for (int g = 0; g < HH; ++g) s[g] *= inv;
  float* op = out + (long)n * EE + h * DD;
  const float* vb = v + (long)n * EE;
#pragma unroll
  for (int d = 0; d < DD; ++d) {
    float o = 0.0f;
#pragma unroll
    for (int g = 0; g < HH; ++g) o += s[g] * vb[g * DD + d];
    op[d] = accum ? (op[d] + o) : o;
  }
}

// ---------------------------------------------------------------------------
extern "C" void kernel_launch(void* const* d_in, const int* in_sizes, int n_in,
                              void* d_out, int out_size, void* d_ws, size_t ws_size,
                              hipStream_t stream) {
  const float* x_node = (const float*)d_in[0];
  const float* x_edge = (const float*)d_in[1];
  const float* adj    = (const float*)d_in[2];
  const float* Wq     = (const float*)d_in[3];
  const float* Wk     = (const float*)d_in[4];
  const float* Wv     = (const float*)d_in[5];
  const float* p_he_h = (const float*)d_in[6];
  const float* p_eh_h = (const float*)d_in[7];
  const float* p_he_e = (const float*)d_in[8];
  const float* p_eh_e = (const float*)d_in[9];
  const float* rms1_h = (const float*)d_in[10];
  const float* rms1_e = (const float*)d_in[11];
  const float* rms2_h = (const float*)d_in[12];
  const float* w1_h   = (const float*)d_in[13];
  const float* b1_h   = (const float*)d_in[14];
  const float* w2_h   = (const float*)d_in[15];
  const float* b2_h   = (const float*)d_in[16];
  const float* w1_e   = (const float*)d_in[17];
  const float* b1_e   = (const float*)d_in[18];
  const float* w2_e   = (const float*)d_in[19];
  const float* b2_e   = (const float*)d_in[20];

  char* wsp = (char*)d_ws;
  auto alloc = [&](size_t b) -> char* {
    char* p = wsp;
    wsp += (b + 255) & ~(size_t)255;
    return p;
  };

  // bf16 operand pool
  bf16* adjb  = (bf16*)alloc((size_t)NN * NN * 2);
  bf16* pehhb = (bf16*)alloc((size_t)NN * NN * 2);
  bf16* phehb = (bf16*)alloc((size_t)NN * NN * 2);
  bf16* peheb = (bf16*)alloc((size_t)NN * NN * 2);
  bf16* pheeb = (bf16*)alloc((size_t)NN * NN * 2);
  bf16* Wqb   = (bf16*)alloc((size_t)8 * EE * EE * 2);
  bf16* Wkb   = (bf16*)alloc((size_t)8 * EE * EE * 2);
  bf16* Wvb   = (bf16*)alloc((size_t)8 * EE * EE * 2);
  bf16* w1hb  = (bf16*)alloc((size_t)FFD * EE * 2);
  bf16* w2hb  = (bf16*)alloc((size_t)EE * FFD * 2);
  bf16* w1eb  = (bf16*)alloc((size_t)FFD * EE * 2);
  bf16* w2eb  = (bf16*)alloc((size_t)EE * FFD * 2);
  bf16* hb    = (bf16*)alloc((size_t)NN * EE * 2);
  bf16* eb    = (bf16*)alloc((size_t)NN * EE * 2);
  bf16* tT    = (bf16*)alloc((size_t)EE * NN * 2);  // transposed q/k panel
  bf16* xb    = (bf16*)alloc((size_t)NN * EE * 2);  // rms2 output
  bf16* ffb   = (bf16*)alloc((size_t)NN * FFD * 2); // FFN hidden (bf16)
  // f32 scratch
  float* qf = (float*)alloc((size_t)NN * EE * 4);
  float* kf = (float*)alloc((size_t)NN * EE * 4);
  float* vf = (float*)alloc((size_t)NN * EE * 4);
  float* xh = (float*)alloc((size_t)NN * EE * 4);
  float* xe = (float*)alloc((size_t)NN * EE * 4);

  auto conv = [&](const float* src, bf16* dst, long n) {
    long n4 = n / 4;
    k_f32_to_bf16<<<dim3((unsigned)((n4 + 255) / 256)), dim3(256), 0, stream>>>(src, dst, n4);
  };
  conv(adj, adjb, (long)NN * NN);
  conv(p_eh_h, pehhb, (long)NN * NN);
  conv(p_he_h, phehb, (long)NN * NN);
  conv(p_eh_e, peheb, (long)NN * NN);
  conv(p_he_e, pheeb, (long)NN * NN);
  conv(Wq, Wqb, (long)8 * EE * EE);
  conv(Wk, Wkb, (long)8 * EE * EE);
  conv(Wv, Wvb, (long)8 * EE * EE);
  conv(w1_h, w1hb, (long)FFD * EE);
  conv(w2_h, w2hb, (long)EE * FFD);
  conv(w1_e, w1eb, (long)FFD * EE);
  conv(w2_e, w2eb, (long)EE * FFD);

  // h = rmsnorm(x_node)*rms1_h ; e = rmsnorm(x_edge)*rms1_e  (bf16)
  k_rmsnorm<<<dim3(NN / 8), dim3(256), 0, stream>>>(x_node, rms1_h, hb);
  k_rmsnorm<<<dim3(NN / 8), dim3(256), 0, stream>>>(x_edge, rms1_e, eb);

  auto gemm_f32 = [&](const bf16* A, int lda, const bf16* Bt, int ldb,
                      int M, int Nc, int K, float* C, int ldc) {
    k_gemm_bt<false, false, true, false, false>
        <<<dim3(Nc / 128, M / 64), dim3(128), 0, stream>>>(
            A, lda, Bt, ldb, K, nullptr, C, nullptr, ldc, nullptr, 0);
  };
  auto gemm_bf16T = [&](const bf16* A, int lda, const bf16* Bt, int ldb,
                        int M, int Nc, int K, bf16* Ct, int ldct) {
    k_gemm_bt<false, false, false, false, true>
        <<<dim3(Nc / 128, M / 64), dim3(128), 0, stream>>>(
            A, lda, Bt, ldb, K, nullptr, nullptr, nullptr, 0, Ct, ldct);
  };
  auto gemm_gelu = [&](const bf16* A, int lda, const bf16* Bt, int ldb,
                       int M, int Nc, int K, const float* bias, bf16* Cb, int ldc) {
    k_gemm_bt<true, true, false, true, false>
        <<<dim3(Nc / 128, M / 64), dim3(128), 0, stream>>>(
            A, lda, Bt, ldb, K, bias, nullptr, Cb, ldc, nullptr, 0);
  };
  auto gemm_bias = [&](const bf16* A, int lda, const bf16* Bt, int ldb,
                       int M, int Nc, int K, const float* bias, float* Cf, int ldc) {
    k_gemm_bt<true, false, true, false, false>
        <<<dim3(Nc / 128, M / 64), dim3(128), 0, stream>>>(
            A, lda, Bt, ldb, K, bias, Cf, nullptr, ldc, nullptr, 0);
  };

  // module order: 0 hh_h, 1 ee_h, 2 eh_h, 3 he_h, 4 hh_e, 5 ee_e, 6 eh_e, 7 he_e
  const bf16* qin[8]  = {hb, eb, eb, hb, hb, eb, eb, hb};
  const bf16* kin[8]  = {hb, eb, hb, eb, hb, eb, hb, eb};
  const int   mode[8] = {1, 0, 2, 2, 1, 0, 2, 2};  // 1=agg q, 2=agg k, 0=none
  const bf16* mat[8]  = {adjb, nullptr, pehhb, phehb, adjb, nullptr, peheb, pheeb};
  float*      accb[8] = {xh, xh, xh, xh, xe, xe, xe, xe};

  for (int m = 0; m < 8; ++m) {
    const bf16* Wqm = Wqb + (long)m * EE * EE;
    const bf16* Wkm = Wkb + (long)m * EE * EE;
    const bf16* Wvm = Wvb + (long)m * EE * EE;
    // v = h @ Wv^T (v_in is always h)
    gemm_f32(hb, EE, Wvm, EE, NN, EE, EE, vf, EE);
    if (mode[m] == 1) {
      // q = mat @ (q_in @ Wq^T): projection emits transposed bf16 panel
      gemm_bf16T(qin[m], EE, Wqm, EE, NN, EE, EE, tT, NN);
      gemm_f32(mat[m], NN, tT, NN, NN, EE, NN, qf, EE);
      gemm_f32(kin[m], EE, Wkm, EE, NN, EE, EE, kf, EE);
    } else if (mode[m] == 2) {
      gemm_f32(qin[m], EE, Wqm, EE, NN, EE, EE, qf, EE);
      gemm_bf16T(kin[m], EE, Wkm, EE, NN, EE, EE, tT, NN);
      gemm_f32(mat[m], NN, tT, NN, NN, EE, NN, kf, EE);
    } else {
      gemm_f32(qin[m], EE, Wqm, EE, NN, EE, EE, qf, EE);
      gemm_f32(kin[m], EE, Wkm, EE, NN, EE, EE, kf, EE);
    }
    const int accum = (m % 4 != 0);  // first module of each branch overwrites
    k_sdpa<<<dim3(NN * HH / 256), dim3(256), 0, stream>>>(qf, kf, vf, accb[m], accum);
  }

  // FFN branch h (rms2_h used for BOTH branches, faithful to reference bug)
  k_rmsnorm<<<dim3(NN / 8), dim3(256), 0, stream>>>(xh, rms2_h, xb);
  gemm_gelu(xb, EE, w1hb, EE, NN, FFD, EE, b1_h, ffb, FFD);
  gemm_bias(ffb, FFD, w2hb, FFD, NN, EE, FFD, b2_h, (float*)d_out, EE);
  // FFN branch e
  k_rmsnorm<<<dim3(NN / 8), dim3(256), 0, stream>>>(xe, rms2_h, xb);
  gemm_gelu(xb, EE, w1eb, EE, NN, FFD, EE, b1_e, ffb, FFD);
  gemm_bias(ffb, FFD, w2eb, FFD, NN, EE, FFD, b2_e, (float*)d_out + (long)NN * EE, EE);

  (void)in_sizes; (void)n_in; (void)out_size; (void)ws_size;
}
